// EncDecConvLSTM_46626164965573
// MI455X (gfx1250) — matrix operand order repaired and checked by reading, not compiled, not run on hardware
//
#include <hip/hip_runtime.h>
#include <hip/hip_bf16.h>

typedef __attribute__((ext_vector_type(16))) _Float16 v16h;
typedef __attribute__((ext_vector_type(8)))  _Float16 v8h;
typedef __attribute__((ext_vector_type(8)))  float    v8f;

// ---------------------------------------------------------------------------
// Fragment index helpers (CDNA5 wave32 WMMA 16x16x32 f16 layouts)
// A (16x32, MxK): lane = (row | hi<<4); halves h: k = h + hi*8 + (h>=8 ? 8 : 0)
// B (32x16, KxN): lane = (n | hi<<4);   halves h: k = h + hi*8 + (h>=8 ? 8 : 0)
// C/D f32 (16x16): lane 0-15: N=lane, M=v; lane 16-31: N=lane-16, M=v+8
// ---------------------------------------------------------------------------
__device__ __forceinline__ int b_k_of(int h, int hi) { return h + hi * 8 + (h >= 8 ? 8 : 0); }

__device__ __forceinline__ float sigm(float x) { return 1.0f / (1.0f + __expf(-x)); }

__device__ __forceinline__ v16h join16(v8h lo, v8h hi) {
    v16h a;
#pragma unroll
    for (int e = 0; e < 8; ++e) { a[e] = lo[e]; a[8 + e] = hi[e]; }
    return a;
}

#define WMMA_F16(A, B, C) \
    __builtin_amdgcn_wmma_f32_16x16x32_f16(false, (A), false, (B), (short)0, (C), false, false)

// Problem constants
#define BATCH 8
#define TSTEPS 12
#define CIN 16
#define HD 96
#define COUT 8
#define CE 112  /* CIN + HD */
#define CD 104  /* COUT + HD */
#define HW 4096 /* 64*64 */
#define MTOT (BATCH * HW) /* 32768 */
#define HSZ (BATCH * HD * HW) /* per-buffer floats for h/c */

// ---------------------------------------------------------------------------
// Weight packing: linear W[K][N] -> B fragments [kIter][nTile][lane][16]
// ---------------------------------------------------------------------------
__global__ void pack_linear_k(const float* __restrict__ W, _Float16* __restrict__ out,
                              int Kact, int Nact, int kIters, int nTiles) {
    int tid = blockIdx.x * blockDim.x + threadIdx.x;
    int total = kIters * nTiles * 32;
    if (tid >= total) return;
    int lane = tid & 31;
    int frag = tid >> 5;
    int nt  = frag % nTiles;
    int kit = frag / nTiles;
    int hi = lane >> 4;
    int n = nt * 16 + (lane & 15);
    _Float16* dst = out + (size_t)frag * 512 + lane * 16;
#pragma unroll
    for (int h = 0; h < 16; ++h) {
        int k = kit * 32 + b_k_of(h, hi);
        float v = (k < Kact && n < Nact) ? W[(size_t)k * Nact + n] : 0.0f;
        dst[h] = (_Float16)v;
    }
}

// Conv weights Kw[O][I][3][3] -> per-tap fragments [rs][kIter][nTile][lane][16]
// B[k=c][n=o] at tap (r,s)
__global__ void pack_conv_k(const float* __restrict__ Kw, _Float16* __restrict__ out,
                            int I, int O, int kIters, int nTiles) {
    int tid = blockIdx.x * blockDim.x + threadIdx.x;
    int total = 9 * kIters * nTiles * 32;
    if (tid >= total) return;
    int lane = tid & 31;
    int frag = tid >> 5;
    int nt  = frag % nTiles;
    int tmp = frag / nTiles;
    int kit = tmp % kIters;
    int rs  = tmp / kIters;
    int r = rs / 3, s = rs % 3;
    int hi = lane >> 4;
    int n = nt * 16 + (lane & 15);
    _Float16* dst = out + (size_t)frag * 512 + lane * 16;
#pragma unroll
    for (int h = 0; h < 16; ++h) {
        int k = kit * 32 + b_k_of(h, hi);
        float v = (k < I && n < O) ? Kw[(((size_t)n * I + k) * 3 + r) * 3 + s] : 0.0f;
        dst[h] = (_Float16)v;
    }
}

__global__ void zero_k(float* p, int n) {
    int i = blockIdx.x * blockDim.x + threadIdx.x;
    if (i < n) p[i] = 0.0f;
}

// ---------------------------------------------------------------------------
// Encoder step: gates = sigmoid/tanh( xc @ W + b ), xc = [x_t ; h] over 112 ch
// Block: 192 thr = 6 waves, one 16-position M-tile; wave w owns Hd channels
// [16w,16w+16). 4 accumulators per wave (f,i,g,o). K = 112 padded to 128.
// ---------------------------------------------------------------------------
__global__ __launch_bounds__(192) void enc_step_k(
    const float* __restrict__ enc_in, float* __restrict__ h, float* __restrict__ c,
    const _Float16* __restrict__ packW, const float* __restrict__ bfp,
    const float* __restrict__ bip, const float* __restrict__ bcp,
    const float* __restrict__ bop, int t) {
    __shared__ alignas(32) _Float16 As[16 * 136];
    const int tid = threadIdx.x;
    const int m0 = blockIdx.x * 16;
    const int b = m0 >> 12;
    const int pos = m0 & 4095;

    for (int idx = tid; idx < 16 * 128; idx += 192) {
        int cc = idx >> 4;
        int j = idx & 15;
        float v = 0.0f;
        if (cc < CIN)
            v = enc_in[((((size_t)b * TSTEPS + t) * CIN + cc) * HW) + pos + j];
        else if (cc < CE)
            v = h[(((size_t)b * HD + (cc - CIN)) * HW) + pos + j];
        As[j * 136 + cc] = (_Float16)v;
    }
    __syncthreads();

    const int wave = tid >> 5, lane = tid & 31, row = lane & 15, hi = lane >> 4;
    const int n = wave * 16 + row;
    v8f acc0 = {}, acc1 = {}, acc2 = {}, acc3 = {};
    const _Float16* arow = As + row * 136;
#pragma unroll
    for (int kit = 0; kit < 4; ++kit) {
        v8h alo = *(const v8h*)(arow + kit * 32 + hi * 8);
        v8h ahi = *(const v8h*)(arow + kit * 32 + 16 + hi * 8);
        v16h a = join16(alo, ahi);
        const _Float16* pb = packW + ((size_t)kit * 6 + wave) * 512 + lane * 16;
        v16h b0 = *(const v16h*)(pb + 0 * 24 * 512);
        v16h b1 = *(const v16h*)(pb + 1 * 24 * 512);
        v16h b2 = *(const v16h*)(pb + 2 * 24 * 512);
        v16h b3 = *(const v16h*)(pb + 3 * 24 * 512);
        acc0 = WMMA_F16(a, b0, acc0);
        acc1 = WMMA_F16(a, b1, acc1);
        acc2 = WMMA_F16(a, b2, acc2);
        acc3 = WMMA_F16(a, b3, acc3);
    }

    const float biasf = bfp[n], biasi = bip[n], biasc = bcp[n], biaso = bop[n];
    const size_t base = ((size_t)b * HD + n) * HW + pos + hi * 8;
    float* cp = c + base;
    float* hp = h + base;
#pragma unroll
    for (int v = 0; v < 8; ++v) {
        float fv = sigm(acc0[v] + biasf);
        float iv = sigm(acc1[v] + biasi);
        float gv = tanhf(acc2[v] + biasc);
        float ov = sigm(acc3[v] + biaso);
        float cn = cp[v] * fv + iv * gv;
        cp[v] = cn;
        hp[v] = tanhf(cn) * ov;
    }
}

// ---------------------------------------------------------------------------
// Decoder step part 1: f,i,g via 3x3 conv (9 shifted K=104 GEMMs), o via
// channel-linear; then c/h update. Reads hOld (halo), writes hNew.
// LDS halo tile Ad[3][18][136] of xc = [dec_x ; hOld].
// ---------------------------------------------------------------------------
__global__ __launch_bounds__(192) void dec_step_k(
    const float* __restrict__ dec_in, const float* __restrict__ hOld,
    float* __restrict__ hNew, float* __restrict__ c,
    const _Float16* __restrict__ packC, const _Float16* __restrict__ packO,
    const float* __restrict__ bfp, const float* __restrict__ bip,
    const float* __restrict__ bcp, const float* __restrict__ bop, int t) {
    __shared__ alignas(32) _Float16 Ad[3 * 18 * 136];
    const int tid = threadIdx.x;
    const int m0 = blockIdx.x * 16;
    const int b = m0 >> 12;
    const int pos = m0 & 4095;
    const int y = pos >> 6;
    const int x0 = pos & 63;

    for (int idx = tid; idx < 3 * 18 * 128; idx += 192) {
        int cc = idx & 127;
        int rem = idx >> 7;
        int xx = rem % 18;
        int r = rem / 18;
        int yy = y + r - 1;
        int x = x0 + xx - 1;
        float v = 0.0f;
        if (yy >= 0 && yy < 64 && x >= 0 && x < 64 && cc < CD) {
            if (cc < COUT)
                v = dec_in[((((size_t)b * TSTEPS + t) * COUT + cc) * HW) + yy * 64 + x];
            else
                v = hOld[(((size_t)b * HD + (cc - COUT)) * HW) + yy * 64 + x];
        }
        Ad[(r * 18 + xx) * 136 + cc] = (_Float16)v;
    }
    __syncthreads();

    const int wave = tid >> 5, lane = tid & 31, row = lane & 15, hi = lane >> 4;
    const int n = wave * 16 + row;
    v8f acc0 = {}, acc1 = {}, acc2 = {}, acc3 = {};
#pragma unroll
    for (int rs = 0; rs < 9; ++rs) {
        int r = rs / 3, s = rs % 3;
        const _Float16* arow = Ad + (r * 18 + row + s) * 136;
#pragma unroll
        for (int kit = 0; kit < 4; ++kit) {
            v8h alo = *(const v8h*)(arow + kit * 32 + hi * 8);
            v8h ahi = *(const v8h*)(arow + kit * 32 + 16 + hi * 8);
            v16h a = join16(alo, ahi);
            const _Float16* pb =
                packC + (((size_t)rs * 4 + kit) * 6 + wave) * 512 + lane * 16;
            v16h bF = *(const v16h*)(pb);
            v16h bI = *(const v16h*)(pb + (size_t)216 * 512);
            v16h bG = *(const v16h*)(pb + (size_t)2 * 216 * 512);
            acc0 = WMMA_F16(a, bF, acc0);
            acc1 = WMMA_F16(a, bI, acc1);
            acc2 = WMMA_F16(a, bG, acc2);
        }
    }
    {  // o gate: channel-linear == center tap of the halo tile
        const _Float16* arow = Ad + (1 * 18 + row + 1) * 136;
#pragma unroll
        for (int kit = 0; kit < 4; ++kit) {
            v8h alo = *(const v8h*)(arow + kit * 32 + hi * 8);
            v8h ahi = *(const v8h*)(arow + kit * 32 + 16 + hi * 8);
            v16h a = join16(alo, ahi);
            v16h bO = *(const v16h*)(packO + ((size_t)kit * 6 + wave) * 512 + lane * 16);
            acc3 = WMMA_F16(a, bO, acc3);
        }
    }

    const float biasf = bfp[n], biasi = bip[n], biasc = bcp[n], biaso = bop[n];
    const size_t base = ((size_t)b * HD + n) * HW + pos + hi * 8;
    float* cp = c + base;
    float* hp = hNew + base;
#pragma unroll
    for (int v = 0; v < 8; ++v) {
        float fv = sigm(acc0[v] + biasf);
        float iv = sigm(acc1[v] + biasi);
        float gv = tanhf(acc2[v] + biasc);
        float ov = sigm(acc3[v] + biaso);
        float cn = cp[v] * fv + iv * gv;
        cp[v] = cn;
        hp[v] = tanhf(cn) * ov;
    }
}

// ---------------------------------------------------------------------------
// Decoder step part 2: rho = conv3x3(hNew, oK) + ob  (96 -> 104 channels),
// then out = rho @ lW + lb (104 -> 8). Block: 224 thr = 7 waves (N=104 pad
// 112); rho staged in LDS, tiny lW projection done VALU-side.
// ---------------------------------------------------------------------------
__global__ __launch_bounds__(224) void rho_out_k(
    const float* __restrict__ hNew, const _Float16* __restrict__ packK,
    const float* __restrict__ ob, const float* __restrict__ lW,
    const float* __restrict__ lb, float* __restrict__ out, int t) {
    __shared__ alignas(32) _Float16 Ah[3 * 18 * 104];
    __shared__ float Rho[16 * 113];
    const int tid = threadIdx.x;
    const int m0 = blockIdx.x * 16;
    const int b = m0 >> 12;
    const int pos = m0 & 4095;
    const int y = pos >> 6;
    const int x0 = pos & 63;

    for (int idx = tid; idx < 3 * 18 * 96; idx += 224) {
        int cc = idx % 96;
        int rem = idx / 96;
        int xx = rem % 18;
        int r = rem / 18;
        int yy = y + r - 1;
        int x = x0 + xx - 1;
        float v = 0.0f;
        if (yy >= 0 && yy < 64 && x >= 0 && x < 64)
            v = hNew[(((size_t)b * HD + cc) * HW) + yy * 64 + x];
        Ah[(r * 18 + xx) * 104 + cc] = (_Float16)v;
    }
    __syncthreads();

    const int wave = tid >> 5, lane = tid & 31, row = lane & 15, hi = lane >> 4;
    const int n = wave * 16 + row;
    v8f acc = {};
#pragma unroll
    for (int rs = 0; rs < 9; ++rs) {
        int r = rs / 3, s = rs % 3;
        const _Float16* arow = Ah + (r * 18 + row + s) * 104;
#pragma unroll
        for (int kit = 0; kit < 3; ++kit) {  // K = 96 = 3*32 exactly
            v8h alo = *(const v8h*)(arow + kit * 32 + hi * 8);
            v8h ahi = *(const v8h*)(arow + kit * 32 + 16 + hi * 8);
            v16h a = join16(alo, ahi);
            v16h bb =
                *(const v16h*)(packK + (((size_t)rs * 3 + kit) * 7 + wave) * 512 + lane * 16);
            acc = WMMA_F16(a, bb, acc);
        }
    }
    if (n < CD) {
        float bias = ob[n];
#pragma unroll
        for (int v = 0; v < 8; ++v) Rho[(hi * 8 + v) * 113 + n] = acc[v] + bias;
    }
    __syncthreads();

    if (tid < 128) {
        int j = tid >> 3;
        int o = tid & 7;
        float s = lb[o];
        for (int cc = 0; cc < CD; ++cc) s += Rho[j * 113 + cc] * lW[cc * COUT + o];
        out[((((size_t)b * TSTEPS + t) * COUT + o) * HW) + pos + j] = s;
    }
}

// ---------------------------------------------------------------------------
extern "C" void kernel_launch(void* const* d_in, const int* in_sizes, int n_in,
                              void* d_out, int out_size, void* d_ws, size_t ws_size,
                              hipStream_t stream) {
    const float* enc_in = (const float*)d_in[0];
    const float* dec_in = (const float*)d_in[1];
    const float* eWf = (const float*)d_in[2];  const float* ebf = (const float*)d_in[3];
    const float* eWi = (const float*)d_in[4];  const float* ebi = (const float*)d_in[5];
    const float* eWc = (const float*)d_in[6];  const float* ebc = (const float*)d_in[7];
    const float* eWo = (const float*)d_in[8];  const float* ebo = (const float*)d_in[9];
    const float* dKf = (const float*)d_in[10]; const float* dbf = (const float*)d_in[11];
    const float* dKi = (const float*)d_in[12]; const float* dbi = (const float*)d_in[13];
    const float* dKc = (const float*)d_in[14]; const float* dbc = (const float*)d_in[15];
    const float* dWo = (const float*)d_in[16]; const float* dbo = (const float*)d_in[17];
    const float* oK  = (const float*)d_in[18]; const float* ob  = (const float*)d_in[19];
    const float* lW  = (const float*)d_in[20]; const float* lb  = (const float*)d_in[21];
    float* out = (float*)d_out;

    char* ws = (char*)d_ws;
    size_t off = 0;
    auto wsAlloc = [&](size_t bytes) -> void* {
        void* p = ws + off;
        off += (bytes + 255) & ~(size_t)255;
        return p;
    };
    float* h0  = (float*)wsAlloc((size_t)HSZ * 4);
    float* h1  = (float*)wsAlloc((size_t)HSZ * 4);
    float* cst = (float*)wsAlloc((size_t)HSZ * 4);
    _Float16* encPack = (_Float16*)wsAlloc((size_t)4 * 24 * 512 * 2);   // 4 gates x 24 frags
    _Float16* decPack = (_Float16*)wsAlloc((size_t)3 * 216 * 512 * 2);  // 3 gates x 216 frags
    _Float16* woPack  = (_Float16*)wsAlloc((size_t)24 * 512 * 2);
    _Float16* okPack  = (_Float16*)wsAlloc((size_t)189 * 512 * 2);

    // h0/h1/cst are contiguous (each HSZ*4 is 256B-aligned): one zero pass.
    zero_k<<<(3 * HSZ + 255) / 256, 256, 0, stream>>>(h0, 3 * HSZ);

    // Pack weights into WMMA B-fragment layout.
    const int encThr = 24 * 32;
    pack_linear_k<<<(encThr + 255) / 256, 256, 0, stream>>>(eWf, encPack + (size_t)0 * 24 * 512, CE, HD, 4, 6);
    pack_linear_k<<<(encThr + 255) / 256, 256, 0, stream>>>(eWi, encPack + (size_t)1 * 24 * 512, CE, HD, 4, 6);
    pack_linear_k<<<(encThr + 255) / 256, 256, 0, stream>>>(eWc, encPack + (size_t)2 * 24 * 512, CE, HD, 4, 6);
    pack_linear_k<<<(encThr + 255) / 256, 256, 0, stream>>>(eWo, encPack + (size_t)3 * 24 * 512, CE, HD, 4, 6);
    const int convThr = 9 * 4 * 6 * 32;
    pack_conv_k<<<(convThr + 255) / 256, 256, 0, stream>>>(dKf, decPack + (size_t)0 * 216 * 512, CD, HD, 4, 6);
    pack_conv_k<<<(convThr + 255) / 256, 256, 0, stream>>>(dKi, decPack + (size_t)1 * 216 * 512, CD, HD, 4, 6);
    pack_conv_k<<<(convThr + 255) / 256, 256, 0, stream>>>(dKc, decPack + (size_t)2 * 216 * 512, CD, HD, 4, 6);
    pack_linear_k<<<(encThr + 255) / 256, 256, 0, stream>>>(dWo, woPack, CD, HD, 4, 6);
    const int okThr = 9 * 3 * 7 * 32;
    pack_conv_k<<<(okThr + 255) / 256, 256, 0, stream>>>(oK, okPack, HD, CD, 3, 7);

    const int nBlocks = MTOT / 16;  // 2048
    for (int t = 0; t < TSTEPS; ++t)
        enc_step_k<<<nBlocks, 192, 0, stream>>>(enc_in, h0, cst, encPack, ebf, ebi, ebc, ebo, t);

    for (int t = 0; t < TSTEPS; ++t) {
        const float* hin = (t & 1) ? h1 : h0;
        float* hout = (t & 1) ? h0 : h1;
        dec_step_k<<<nBlocks, 192, 0, stream>>>(dec_in, hin, hout, cst, decPack, woPack,
                                                dbf, dbi, dbc, dbo, t);
        rho_out_k<<<nBlocks, 224, 0, stream>>>(hout, okPack, ob, lW, lb, out, t);
    }
}
